// ParallelRetention_81381040325184
// MI455X (gfx1250) — compile-verified, matrix operand order checked by reading, not run on hardware
//
#include <hip/hip_runtime.h>
#include <hip/hip_bf16.h>

// ---------- types ----------
typedef __attribute__((ext_vector_type(16))) __bf16        v16bf;
typedef __attribute__((ext_vector_type(8)))  float         v8f;
typedef __attribute__((ext_vector_type(4)))  float         f32x4;
typedef __attribute__((ext_vector_type(4)))  unsigned int  u32x4;
typedef __attribute__((ext_vector_type(2)))  unsigned int  u32x2;
typedef __attribute__((ext_vector_type(4)))  unsigned int  v4u;
typedef __attribute__((ext_vector_type(8)))  int           v8i;
typedef __attribute__((ext_vector_type(4)))  int           v4i;

#define T_DIM   64
#define C_DIM   2048
#define B_DIM   32
#define M_TOT   2048      // B*T

// LDS tile: 128 rows x 64 bf16; row pitch 72 ushort = 144B.
// 144B is 16B-aligned (ds b128 ok) and 36*row mod 64 banks are distinct for rows 0..15.
#define LROW    72

#if __has_builtin(__builtin_amdgcn_tensor_load_to_lds)
#define HAVE_TDM 1
#else
#define HAVE_TDM 0
#endif

union BF16Frag {
    v16bf v;
    u32x4 q[2];
    u32x2 d[4];
};

__device__ __forceinline__ unsigned short f2bf(float f) {
    unsigned int u = __float_as_uint(f);
    unsigned int r = u + 0x7FFFu + ((u >> 16) & 1u);   // round-to-nearest-even
    return (unsigned short)(r >> 16);
}

// A/B fragment from global, row-major [rows, ld] bf16, 16B-aligned rows.
// lane 0-15: row=lane, K 0..7 & 16..23 ; lane 16-31: row=lane-16, K 8..15 & 24..31
__device__ __forceinline__ void load_frag_g(BF16Frag& f, const unsigned short* p0,
                                            int ld, int lane) {
    const unsigned short* p = p0 + (lane & 15) * ld + ((lane >> 4) << 3);
    f.q[0] = *(const u32x4*)(p);
    f.q[1] = *(const u32x4*)(p + 16);
}

// Fragment from LDS tile (pitch LROW ushorts, 16B-aligned): k-substep ks in {0,32}
__device__ __forceinline__ void load_frag_lds(BF16Frag& f, const unsigned short* s,
                                              int row0, int ks, int lane) {
    const unsigned short* p = s + (row0 + (lane & 15)) * LROW + ks + ((lane >> 4) << 3);
    f.q[0] = *(const u32x4*)(p);
    f.q[1] = *(const u32x4*)(p + 16);
}

__device__ __forceinline__ v8f wmma_bf16(const BF16Frag& a, const BF16Frag& b, v8f c) {
    return __builtin_amdgcn_wmma_f32_16x16x32_bf16(false, a.v, false, b.v,
                                                   (short)0, c, false, false);
}

#if HAVE_TDM
// Issue one TDM tile load: 128 rows x 64 bf16 (viewed as 32 x 128 dwords),
// global row pitch = C_DIM bf16 = 1024 dwords; LDS pad: +4 dwords every 32 dwords
// (pad_interval code 4 = 32 dwords, pad_amount code 3 = 4 dwords) -> 144B LDS pitch.
__device__ __forceinline__ void tdm_load_tile(const unsigned short* gsrc,
                                              unsigned int lds_off) {
    unsigned long long ga = (unsigned long long)(size_t)(const void*)gsrc;
    const unsigned td0 = 1u << 20, td1 = 1u << 20;   // generous OOB dims
    v4u g0;
    g0[0] = 1u;                                        // count=1, user mode
    g0[1] = lds_off;                                   // lds_addr (bytes)
    g0[2] = (unsigned)(ga & 0xFFFFFFFFu);              // global_addr[31:0]
    g0[3] = (unsigned)((ga >> 32) & 0x01FFFFFFu) | (2u << 30);  // addr[56:32] | type=2
    v8i g1;
    g1[0] = (int)((2u << 16) | (1u << 20) | (4u << 22) | (3u << 25)); // data_size=4B, pad
    g1[1] = (int)((td0 & 0xFFFFu) << 16);                             // tensor_dim0 lo16
    g1[2] = (int)((td0 >> 16) | ((td1 & 0xFFFFu) << 16));             // dim0 hi | dim1 lo
    g1[3] = (int)((td1 >> 16) | (32u << 16));                         // dim1 hi | tile_dim0=32
    g1[4] = (int)(128u);                                              // tile_dim1=128
    g1[5] = (int)(1024u);                                             // dim0_stride lo32
    g1[6] = 0;
    g1[7] = 0;
    v4i gz = {0, 0, 0, 0};
#if __clang_major__ >= 23
    v8i gz8 = {0, 0, 0, 0, 0, 0, 0, 0};
    __builtin_amdgcn_tensor_load_to_lds(g0, g1, gz, gz, gz8, 0);
#else
    __builtin_amdgcn_tensor_load_to_lds(g0, g1, gz, gz, 0);
#endif
}
#endif

// ---------- f32 -> bf16 conversion ----------
__global__ __launch_bounds__(256) void cvt_f32_bf16(const float* __restrict__ src,
                                                    unsigned short* __restrict__ dst,
                                                    int n) {
    int i = (blockIdx.x * 256 + threadIdx.x) * 4;
    if (i < n) {
        f32x4 x = *(const f32x4*)(src + i);
        u32x2 o;
        o[0] = (unsigned int)f2bf(x[0]) | ((unsigned int)f2bf(x[1]) << 16);
        o[1] = (unsigned int)f2bf(x[2]) | ((unsigned int)f2bf(x[3]) << 16);
        *(u32x2*)(dst + i) = o;
    }
}

// ---------- main 2048x2048x2048 bf16 GEMM:  D = A * Bw^T + bias ----------
// MODE 0: bf16 row-major [M, C] (+bias)                 (Q, K)
// MODE 1: bf16 transposed per batch [b][C][T] (+bias)   (V)
// MODE 2: f32 row-major, +bias, PReLU                   (final)
template<int MODE>
__global__ __launch_bounds__(256) void gemm_bf16(const unsigned short* __restrict__ A,
                                                 const unsigned short* __restrict__ Bw,
                                                 const float* __restrict__ bias,
                                                 void* __restrict__ dst,
                                                 const float* __restrict__ prelu) {
    __shared__ unsigned short sA[128 * LROW];
    __shared__ unsigned short sB[128 * LROW];

    const int tid  = threadIdx.x;
    const int lane = tid & 31;
    const int wave = tid >> 5;          // 0..7
    const int wm   = wave >> 2;         // 0..1  -> 64 rows
    const int wn   = wave & 3;          // 0..3  -> 32 cols
    const int m0   = blockIdx.y * 128;
    const int n0   = blockIdx.x * 128;

    float pa = 0.0f;
    if (MODE == 2) pa = *prelu;

#if HAVE_TDM
    const unsigned ldsA = (unsigned)(size_t)(void*)sA;
    const unsigned ldsB = (unsigned)(size_t)(void*)sB;
#endif

    v8f acc[4][2];
    const v8f vzero = {0.f, 0.f, 0.f, 0.f, 0.f, 0.f, 0.f, 0.f};
#pragma unroll
    for (int mt = 0; mt < 4; ++mt)
#pragma unroll
        for (int nt = 0; nt < 2; ++nt) acc[mt][nt] = vzero;

    for (int k0 = 0; k0 < C_DIM; k0 += 64) {
#if HAVE_TDM
        if (wave == 0) {
            tdm_load_tile(A  + (size_t)m0 * C_DIM + k0, ldsA);
            tdm_load_tile(Bw + (size_t)n0 * C_DIM + k0, ldsB);
            __builtin_amdgcn_s_wait_tensorcnt(0);
        }
        if (wave == 7 && k0 + 64 < C_DIM) {   // warm GL2 for the next K-tile
            __builtin_prefetch(A  + (size_t)(m0 + (lane << 2)) * C_DIM + k0 + 64, 0, 1);
            __builtin_prefetch(Bw + (size_t)(n0 + (lane << 2)) * C_DIM + k0 + 64, 0, 1);
        }
#else
        // manual copy into the same padded layout: 1024 chunks of 8 bf16 per matrix
#pragma unroll
        for (int L = tid; L < 1024; L += 256) {
            int row = L >> 3;
            int col = (L & 7) << 3;
            u32x4 av = *(const u32x4*)(A  + (size_t)(m0 + row) * C_DIM + k0 + col);
            u32x4 bv = *(const u32x4*)(Bw + (size_t)(n0 + row) * C_DIM + k0 + col);
            *(u32x4*)(sA + row * LROW + col) = av;
            *(u32x4*)(sB + row * LROW + col) = bv;
        }
#endif
        __syncthreads();

#pragma unroll
        for (int ks = 0; ks < 64; ks += 32) {
            BF16Frag af[4], bf[2];
#pragma unroll
            for (int mt = 0; mt < 4; ++mt)
                load_frag_lds(af[mt], sA, wm * 64 + mt * 16, ks, lane);
#pragma unroll
            for (int nt = 0; nt < 2; ++nt)
                load_frag_lds(bf[nt], sB, wn * 32 + nt * 16, ks, lane);
#pragma unroll
            for (int mt = 0; mt < 4; ++mt)
#pragma unroll
                for (int nt = 0; nt < 2; ++nt)
                    acc[mt][nt] = wmma_bf16(af[mt], bf[nt], acc[mt][nt]);
        }
        __syncthreads();
    }

    // epilogue
    const int half  = lane >> 4;
    const int nlane = lane & 15;
#pragma unroll
    for (int mt = 0; mt < 4; ++mt) {
#pragma unroll
        for (int nt = 0; nt < 2; ++nt) {
            int n    = n0 + wn * 32 + nt * 16 + nlane;
            float bs = bias[n];
            int mbase = m0 + wm * 64 + mt * 16 + half * 8;
#pragma unroll
            for (int r = 0; r < 8; ++r) {
                int m = mbase + r;
                float val = acc[mt][nt][r] + bs;
                if (MODE == 0) {
                    ((unsigned short*)dst)[(size_t)m * C_DIM + n] = f2bf(val);
                } else if (MODE == 1) {
                    int bi = m >> 6, t = m & 63;
                    ((unsigned short*)dst)[(size_t)bi * C_DIM * T_DIM + (size_t)n * T_DIM + t] = f2bf(val);
                } else {
                    val = (val >= 0.0f) ? val : pa * val;
                    ((float*)dst)[(size_t)m * C_DIM + n] = val;
                }
            }
        }
    }
}

// ---------- scores: S[b] = (q[b] k[b]^T) * d_gamma, bf16 out [b][64][64] ----------
__global__ __launch_bounds__(128) void scores_kernel(const unsigned short* __restrict__ q,
                                                     const unsigned short* __restrict__ k,
                                                     const float* __restrict__ gamma,
                                                     unsigned short* __restrict__ S) {
    const int b    = blockIdx.x;
    const int lane = threadIdx.x & 31;
    const int wave = threadIdx.x >> 5;   // 0..3 -> 16 rows of M
    const unsigned short* qb = q + (size_t)b * T_DIM * C_DIM;
    const unsigned short* kb = k + (size_t)b * T_DIM * C_DIM;

    const v8f vzero = {0.f, 0.f, 0.f, 0.f, 0.f, 0.f, 0.f, 0.f};
    v8f acc[4];
#pragma unroll
    for (int nt = 0; nt < 4; ++nt) acc[nt] = vzero;

    for (int k0 = 0; k0 < C_DIM; k0 += 32) {
        BF16Frag af;
        load_frag_g(af, qb + (size_t)wave * 16 * C_DIM + k0, C_DIM, lane);
#pragma unroll
        for (int nt = 0; nt < 4; ++nt) {
            BF16Frag bf;
            load_frag_g(bf, kb + (size_t)nt * 16 * C_DIM + k0, C_DIM, lane);
            acc[nt] = wmma_bf16(af, bf, acc[nt]);
        }
    }
    const int half  = lane >> 4;
    const int nlane = lane & 15;
#pragma unroll
    for (int nt = 0; nt < 4; ++nt) {
        int n = nt * 16 + nlane;
#pragma unroll
        for (int r = 0; r < 8; ++r) {
            int m = wave * 16 + half * 8 + r;
            float val = acc[nt][r] * gamma[m * T_DIM + n];
            S[(size_t)b * T_DIM * T_DIM + (size_t)m * T_DIM + n] = f2bf(val);
        }
    }
}

// ---------- ret[b] = S[b] @ v[b] ; vT layout [b][C][T], out f32 [b][C][T] ----------
__global__ __launch_bounds__(128) void ret_kernel(const unsigned short* __restrict__ S,
                                                  const unsigned short* __restrict__ vT,
                                                  float* __restrict__ ret) {
    const int b    = blockIdx.y;
    const int n0   = blockIdx.x * 128 + (threadIdx.x >> 5) * 32;
    const int lane = threadIdx.x & 31;
    const unsigned short* Sb = S  + (size_t)b * T_DIM * T_DIM;
    const unsigned short* vb = vT + (size_t)b * C_DIM * T_DIM;

    const v8f vzero = {0.f, 0.f, 0.f, 0.f, 0.f, 0.f, 0.f, 0.f};
    v8f acc[4][2];
#pragma unroll
    for (int mt = 0; mt < 4; ++mt)
#pragma unroll
        for (int nt = 0; nt < 2; ++nt) acc[mt][nt] = vzero;

#pragma unroll
    for (int k0 = 0; k0 < T_DIM; k0 += 32) {
        BF16Frag af[4], bf[2];
#pragma unroll
        for (int mt = 0; mt < 4; ++mt)
            load_frag_g(af[mt], Sb + (size_t)mt * 16 * T_DIM + k0, T_DIM, lane);
#pragma unroll
        for (int nt = 0; nt < 2; ++nt)
            load_frag_g(bf[nt], vb + (size_t)(n0 + nt * 16) * T_DIM + k0, T_DIM, lane);
#pragma unroll
        for (int mt = 0; mt < 4; ++mt)
#pragma unroll
            for (int nt = 0; nt < 2; ++nt)
                acc[mt][nt] = wmma_bf16(af[mt], bf[nt], acc[mt][nt]);
    }
    const int half  = lane >> 4;
    const int nlane = lane & 15;
#pragma unroll
    for (int mt = 0; mt < 4; ++mt) {
#pragma unroll
        for (int nt = 0; nt < 2; ++nt) {
            int n = n0 + nt * 16 + nlane;
            int mbase = mt * 16 + half * 8;
            float* p = ret + (size_t)b * C_DIM * T_DIM + (size_t)n * T_DIM + mbase;
            *(f32x4*)(p)     = ((const f32x4*)&acc[mt][nt])[0];
            *(f32x4*)(p + 4) = ((const f32x4*)&acc[mt][nt])[1];
        }
    }
}

// ---------- GroupNorm over [64ch x 64t] contiguous per (b,g); y bf16 [b][T][C] ----------
__global__ __launch_bounds__(256) void gn_kernel(const float* __restrict__ ret,
                                                 const float* __restrict__ gw,
                                                 const float* __restrict__ gb,
                                                 unsigned short* __restrict__ y) {
    const int b = blockIdx.x >> 5;
    const int g = blockIdx.x & 31;
    const int tid = threadIdx.x;
    const float* base = ret + (size_t)b * C_DIM * T_DIM + (size_t)g * 64 * T_DIM;

    float xv[16];
    float s = 0.f, sq = 0.f;
    const f32x4* src = (const f32x4*)(base + tid * 16);
#pragma unroll
    for (int j = 0; j < 4; ++j) {
        f32x4 v = src[j];
        ((f32x4*)xv)[j] = v;
#pragma unroll
        for (int e = 0; e < 4; ++e) { s += v[e]; sq += v[e] * v[e]; }
    }
    __shared__ float red[512];
    red[tid] = s; red[256 + tid] = sq;
    __syncthreads();
    for (int off = 128; off > 0; off >>= 1) {
        if (tid < off) { red[tid] += red[tid + off]; red[256 + tid] += red[256 + tid + off]; }
        __syncthreads();
    }
    float mean = red[0] * (1.0f / 4096.0f);
    float var  = red[256] * (1.0f / 4096.0f) - mean * mean;
    float rstd = rsqrtf(var + 1e-5f);

    int cl = (tid * 16) >> 6;
    int t0 = (tid * 16) & 63;
    int c  = g * 64 + cl;
    float w  = gw[c] * rstd;
    float bb = gb[c];
    unsigned short* dst = y + (size_t)b * T_DIM * C_DIM + c;
#pragma unroll
    for (int j = 0; j < 16; ++j) {
        float val = (xv[j] - mean) * w + bb;
        dst[(size_t)(t0 + j) * C_DIM] = f2bf(val);
    }
}

// ---------- launch ----------
extern "C" void kernel_launch(void* const* d_in, const int* in_sizes, int n_in,
                              void* d_out, int out_size, void* d_ws, size_t ws_size,
                              hipStream_t stream) {
    const float* x   = (const float*)d_in[0];
    const float* dg  = (const float*)d_in[1];
    const float* Wq  = (const float*)d_in[2];
    const float* bq  = (const float*)d_in[3];
    const float* Wk  = (const float*)d_in[4];
    const float* bk  = (const float*)d_in[5];
    const float* Wv  = (const float*)d_in[6];
    const float* bv  = (const float*)d_in[7];
    const float* Wo  = (const float*)d_in[8];
    const float* bo  = (const float*)d_in[9];
    const float* gnw = (const float*)d_in[10];
    const float* gnb = (const float*)d_in[11];
    const float* pa  = (const float*)d_in[12];

    char* ws = (char*)d_ws;
    const size_t MB = 1u << 20;
    unsigned short* xb  = (unsigned short*)(ws);
    unsigned short* Wqb = (unsigned short*)(ws + 8  * MB);
    unsigned short* Wkb = (unsigned short*)(ws + 16 * MB);
    unsigned short* Wvb = (unsigned short*)(ws + 24 * MB);
    unsigned short* Wob = (unsigned short*)(ws + 32 * MB);
    unsigned short* qb  = (unsigned short*)(ws + 40 * MB);
    unsigned short* kb  = (unsigned short*)(ws + 48 * MB);
    unsigned short* vTb = (unsigned short*)(ws + 56 * MB);
    unsigned short* Sb  = (unsigned short*)(ws + 64 * MB);
    float*          rtb = (float*)         (ws + 65 * MB);
    unsigned short* yb  = (unsigned short*)(ws + 81 * MB);

    const int NELT = M_TOT * C_DIM;          // 4,194,304
    const int CVT_BLOCKS = NELT / (256 * 4); // 4096

    cvt_f32_bf16<<<CVT_BLOCKS, 256, 0, stream>>>(x,  xb,  NELT);
    cvt_f32_bf16<<<CVT_BLOCKS, 256, 0, stream>>>(Wq, Wqb, NELT);
    cvt_f32_bf16<<<CVT_BLOCKS, 256, 0, stream>>>(Wk, Wkb, NELT);
    cvt_f32_bf16<<<CVT_BLOCKS, 256, 0, stream>>>(Wv, Wvb, NELT);
    cvt_f32_bf16<<<CVT_BLOCKS, 256, 0, stream>>>(Wo, Wob, NELT);

    dim3 gg(16, 16);
    gemm_bf16<0><<<gg, 256, 0, stream>>>(xb, Wqb, bq, (void*)qb,  nullptr);
    gemm_bf16<0><<<gg, 256, 0, stream>>>(xb, Wkb, bk, (void*)kb,  nullptr);
    gemm_bf16<1><<<gg, 256, 0, stream>>>(xb, Wvb, bv, (void*)vTb, nullptr);

    scores_kernel<<<B_DIM, 128, 0, stream>>>(qb, kb, dg, Sb);
    ret_kernel<<<dim3(16, B_DIM), 128, 0, stream>>>(Sb, vTb, rtb);
    gn_kernel<<<B_DIM * 32, 256, 0, stream>>>(rtb, gnw, gnb, yb);

    gemm_bf16<2><<<gg, 256, 0, stream>>>(yb, Wob, bo, d_out, pa);
}
